// Sep_Autoencoder_13005160972994
// MI455X (gfx1250) — matrix operand order repaired and checked by reading, not compiled, not run on hardware
//
#include <hip/hip_runtime.h>
#include <hip/hip_bf16.h>
#include <math.h>

#define B_  256
#define T_  512
#define BT_ 131072      // B_*T_
#define NM  1024        // codebook size
#define FCK 16384       // T_*32 features into FC
#define FCO 512
#define SPLITK 8

typedef __attribute__((ext_vector_type(16))) _Float16 v16h;
typedef __attribute__((ext_vector_type(8)))  _Float16 v8h;
typedef __attribute__((ext_vector_type(8)))  float    v8f;

// A-matrix fragment load (16x32 f16 A, this code's k order, not the ISA table order):
//   e0..15 -> K=khi*16+e (contiguous)
// caller passes p = rowBase + khi*16
__device__ __forceinline__ v16h load_afrag(const _Float16* p) {
    v8h lo = *(const v8h*)(p);
    v8h hi = *(const v8h*)(p + 8);
    return __builtin_shufflevector(lo, hi, 0,1,2,3,4,5,6,7,8,9,10,11,12,13,14,15);
}

// ---------------------------------------------------------------------------
// Encoder conv 1: (B,1,T) -> (B,5,T), k=3, SAME, relu.  Output layout (bt, 5)
// ---------------------------------------------------------------------------
__global__ void enc_conv1(const float* __restrict__ x, const float* __restrict__ w,
                          const float* __restrict__ bias, float* __restrict__ y) {
    int bt = blockIdx.x * blockDim.x + threadIdx.x;
    if (bt >= BT_) return;
    int t = bt & (T_ - 1);
    float in[3];
#pragma unroll
    for (int k = 0; k < 3; ++k) {
        int tt = t + k - 1;
        in[k] = ((unsigned)tt < T_) ? x[bt + k - 1] : 0.f;
    }
#pragma unroll
    for (int co = 0; co < 5; ++co) {
        float s = bias[co];
#pragma unroll
        for (int k = 0; k < 3; ++k) s = fmaf(w[co * 3 + k], in[k], s);
        y[(size_t)bt * 5 + co] = fmaxf(s, 0.f);
    }
}

// ---------------------------------------------------------------------------
// Encoder conv 2: (B,5,T) -> (B,10,T), k=5, SAME, relu.  (bt,5) -> (bt,10)
// ---------------------------------------------------------------------------
__global__ void enc_conv2(const float* __restrict__ h1, const float* __restrict__ w,
                          const float* __restrict__ bias, float* __restrict__ y) {
    int bt = blockIdx.x * blockDim.x + threadIdx.x;
    if (bt >= BT_) return;
    int t = bt & (T_ - 1);
    float acc[10];
#pragma unroll
    for (int co = 0; co < 10; ++co) acc[co] = bias[co];
    for (int k = 0; k < 5; ++k) {
        int tt = t + k - 2;
        if ((unsigned)tt < T_) {
            const float* hp = h1 + (size_t)(bt + k - 2) * 5;
            for (int ci = 0; ci < 5; ++ci) {
                float v = hp[ci];
#pragma unroll
                for (int co = 0; co < 10; ++co)
                    acc[co] = fmaf(w[(co * 5 + ci) * 5 + k], v, acc[co]);
            }
        }
    }
#pragma unroll
    for (int co = 0; co < 10; ++co) y[(size_t)bt * 10 + co] = fmaxf(acc[co], 0.f);
}

// ---------------------------------------------------------------------------
// Encoder conv 3: (B,10,T) -> (B,64,T), k=7, SAME, tanh -> xph (bt,64) f16
// (no row norms: softmax is invariant to per-row constants)
// ---------------------------------------------------------------------------
__global__ void enc_conv3(const float* __restrict__ h2, const float* __restrict__ w,
                          const float* __restrict__ bias, _Float16* __restrict__ xph) {
    int bt = blockIdx.x * blockDim.x + threadIdx.x;
    if (bt >= BT_) return;
    int t = bt & (T_ - 1);
    float acc[64];
#pragma unroll
    for (int co = 0; co < 64; ++co) acc[co] = bias[co];
    for (int k = 0; k < 7; ++k) {
        int tt = t + k - 3;
        if ((unsigned)tt < T_) {
            const float* hp = h2 + (size_t)(bt + k - 3) * 10;
            for (int ci = 0; ci < 10; ++ci) {
                float v = hp[ci];
#pragma unroll
                for (int co = 0; co < 64; ++co)
                    acc[co] = fmaf(w[(co * 10 + ci) * 7 + k], v, acc[co]);
            }
        }
    }
#pragma unroll
    for (int co = 0; co < 64; ++co)
        xph[(size_t)bt * 64 + co] = (_Float16)tanhf(acc[co]);
}

// ---------------------------------------------------------------------------
// Pre-swizzle codebook into WMMA B-fragment order + column norms.
// B k order used here: lo lanes hold K=0..15 (=e), hi lanes K=16..31 (=16+e); lane&15 = N
// ---------------------------------------------------------------------------
__global__ void prep_code(const float* __restrict__ code,
                          _Float16* __restrict__ codeA, _Float16* __restrict__ codeB,
                          float* __restrict__ cnorm) {
    int tid = blockIdx.x * blockDim.x + threadIdx.x;
    if (tid < NM) {
        float s = 0.f;
        for (int c = 0; c < 32; ++c) { float v = code[c * NM + tid]; s += v * v; }
        cnorm[tid] = s;
    }
    if (tid < 2048) {                       // dist-GEMM B frags: B[k][n] = code[k][n]
        int lane = tid & 31, mt = tid >> 5;
        int khi = lane >> 4, n = mt * 16 + (lane & 15);
#pragma unroll
        for (int e = 0; e < 16; ++e)
            codeA[(size_t)tid * 16 + e] = (_Float16)code[(e + 16 * khi) * NM + n];
    } else if (tid < 4096) {                // P@code^T B frags: B[k][c] = code[c][k]
        int s2 = tid - 2048;
        int lane = s2 & 31, kc = (s2 >> 5) & 31, nt = s2 >> 10;
        int khi = lane >> 4, c = nt * 16 + (lane & 15);
#pragma unroll
        for (int e = 0; e < 16; ++e)
            codeB[(size_t)s2 * 16 + e] =
                (_Float16)code[c * NM + kc * 32 + e + 16 * khi];
    }
}

// ---------------------------------------------------------------------------
// fc_w fp32 -> f16
// ---------------------------------------------------------------------------
__global__ void cvt_f16(const float* __restrict__ src, _Float16* __restrict__ dst, int n) {
    int i = blockIdx.x * blockDim.x + threadIdx.x;
    if (i < n) dst[i] = (_Float16)src[i];
}

// ---------------------------------------------------------------------------
// Fused soft VQ: dist GEMM (WMMA) -> streaming softmax -> P@code^T (WMMA).
// One wave per 16-row tile; blockIdx.y selects channel half.
//  - row max tracked in registers during pass A, reduced with shfl_xor
//  - exp pass vectorized (v8h <-> ds b128), stores UNnormalized exp
//  - 1/sum folded into the pass-B epilogue
// ---------------------------------------------------------------------------
__global__ __launch_bounds__(32) void quantize_kernel(
        const _Float16* __restrict__ xph,
        const _Float16* __restrict__ codeA, const _Float16* __restrict__ codeB,
        const float* __restrict__ cnorm,
        float* __restrict__ xq1, float* __restrict__ xq2) {
    const int lane = threadIdx.x;
    const int r = lane & 15, khi = lane >> 4;
    const int half = blockIdx.y;
    const size_t rbase = (size_t)blockIdx.x * 16;
    float* __restrict__ xq = half ? xq2 : xq1;

    __shared__ _Float16 pbuf[16][NM];   // 32 KB: z scores, then exp(z - zmax) (f16)
    __shared__ float red[32];
    __shared__ float rmax16[16];
    __shared__ float inv16[16];

    // A fragment: 16 rows x K=32 channels of this half (loop invariant)
    const v16h a = load_afrag(xph + (rbase + r) * 64 + half * 32 + khi * 16);

    // ---- pass A: z = 2*x.c - |c|^2 for all 1024 codes; track row maxima ----
    float zmax8[8];
#pragma unroll
    for (int v = 0; v < 8; ++v) zmax8[v] = -1e30f;

    for (int mt = 0; mt < 64; ++mt) {
        v16h b = *(const v16h*)(codeA + ((size_t)mt * 32 + lane) * 16);
        v8f s = {};
        s = __builtin_amdgcn_wmma_f32_16x16x32_f16(false, a, false, b,
                                                   (short)0, s, false, false);
        float cn = cnorm[mt * 16 + r];
#pragma unroll
        for (int v = 0; v < 8; ++v) {
            float z = 2.f * s[v] - cn;
            zmax8[v] = fmaxf(zmax8[v], z);
            pbuf[v + khi * 8][mt * 16 + r] = (_Float16)z;
        }
    }
    // reduce maxima across the 16 lanes of each half-group
#pragma unroll
    for (int v = 0; v < 8; ++v) {
        float m = zmax8[v];
        m = fmaxf(m, __shfl_xor(m, 1, 32));
        m = fmaxf(m, __shfl_xor(m, 2, 32));
        m = fmaxf(m, __shfl_xor(m, 4, 32));
        m = fmaxf(m, __shfl_xor(m, 8, 32));
        zmax8[v] = m;
    }
    if (r == 0) {                       // lanes 0 & 16 publish rows khi*8..khi*8+7
#pragma unroll
        for (int v = 0; v < 8; ++v) rmax16[khi * 8 + v] = zmax8[v];
    }
    __syncthreads();

    // ---- exp + sum: lane (r,khi) owns row r, cols khi*512.. (vectorized) ----
    {
        float zm = rmax16[r];
        float sum = 0.f;
        _Float16* prow = &pbuf[r][khi * 512];
        for (int c = 0; c < 64; ++c) {
            v8h z8 = *(v8h*)(prow + c * 8);
            v8h e8;
#pragma unroll
            for (int e = 0; e < 8; ++e) {
                float ev = __expf((float)z8[e] - zm);
                sum += ev;
                e8[e] = (_Float16)ev;
            }
            *(v8h*)(prow + c * 8) = e8;
        }
        red[lane] = sum;
        __syncthreads();
        if (lane < 16) inv16[lane] = 1.f / (red[lane] + red[lane + 16]);
    }
    __syncthreads();

    float inv8[8];
#pragma unroll
    for (int v = 0; v < 8; ++v) inv8[v] = inv16[v + khi * 8];

    // ---- pass B: xq(16x32) = (expP @ code^T) * inv_row ----
#pragma unroll
    for (int nt = 0; nt < 2; ++nt) {
        v8f acc = {};
        for (int kc = 0; kc < 32; ++kc) {
            v16h pa = load_afrag(&pbuf[r][kc * 32 + khi * 16]);
            v16h pb = *(const v16h*)(codeB + (((size_t)nt * 32 + kc) * 32 + lane) * 16);
            acc = __builtin_amdgcn_wmma_f32_16x16x32_f16(false, pa, false, pb,
                                                         (short)0, acc, false, false);
        }
#pragma unroll
        for (int v = 0; v < 8; ++v)
            xq[(rbase + v + khi * 8) * 32 + nt * 16 + r] = acc[v] * inv8[v];
    }
}

// ---------------------------------------------------------------------------
// Decoder convs (blockIdx.y = half)
// ---------------------------------------------------------------------------
__global__ void dec_conv1(const float* __restrict__ xq1, const float* __restrict__ xq2,
                          const float* __restrict__ w1, const float* __restrict__ w2,
                          const float* __restrict__ b1, const float* __restrict__ b2,
                          float* __restrict__ d1) {
    int bt = blockIdx.x * blockDim.x + threadIdx.x;
    if (bt >= BT_) return;
    int half = blockIdx.y;
    const float* xq = half ? xq2 : xq1;
    const float* w  = half ? w2 : w1;
    const float* bs = half ? b2 : b1;
    float* o = d1 + (size_t)half * BT_ * 5;
    int t = bt & (T_ - 1);
    float acc[5];
#pragma unroll
    for (int co = 0; co < 5; ++co) acc[co] = bs[co];
    for (int k = 0; k < 7; ++k) {
        int tt = t + k - 3;
        if ((unsigned)tt < T_) {
            const float* xp = xq + (size_t)(bt + k - 3) * 32;
            for (int ci = 0; ci < 32; ++ci) {
                float v = xp[ci];
#pragma unroll
                for (int co = 0; co < 5; ++co)
                    acc[co] = fmaf(w[(co * 32 + ci) * 7 + k], v, acc[co]);
            }
        }
    }
#pragma unroll
    for (int co = 0; co < 5; ++co) o[(size_t)bt * 5 + co] = fmaxf(acc[co], 0.f);
}

__global__ void dec_conv2(const float* __restrict__ d1,
                          const float* __restrict__ w1, const float* __restrict__ w2,
                          const float* __restrict__ b1, const float* __restrict__ b2,
                          float* __restrict__ d2) {
    int bt = blockIdx.x * blockDim.x + threadIdx.x;
    if (bt >= BT_) return;
    int half = blockIdx.y;
    const float* in = d1 + (size_t)half * BT_ * 5;
    const float* w  = half ? w2 : w1;
    const float* bs = half ? b2 : b1;
    float* o = d2 + (size_t)half * BT_ * 10;
    int t = bt & (T_ - 1);
    float acc[10];
#pragma unroll
    for (int co = 0; co < 10; ++co) acc[co] = bs[co];
    for (int k = 0; k < 5; ++k) {
        int tt = t + k - 2;
        if ((unsigned)tt < T_) {
            const float* hp = in + (size_t)(bt + k - 2) * 5;
            for (int ci = 0; ci < 5; ++ci) {
                float v = hp[ci];
#pragma unroll
                for (int co = 0; co < 10; ++co)
                    acc[co] = fmaf(w[(co * 5 + ci) * 5 + k], v, acc[co]);
            }
        }
    }
#pragma unroll
    for (int co = 0; co < 10; ++co) o[(size_t)bt * 10 + co] = fmaxf(acc[co], 0.f);
}

// conv3 (10->32, k=3) + relu, emit f16 in FC layout: fcin[b][co*T + t]
__global__ void dec_conv3(const float* __restrict__ d2,
                          const float* __restrict__ w1, const float* __restrict__ w2,
                          const float* __restrict__ b1, const float* __restrict__ b2,
                          _Float16* __restrict__ fcin1, _Float16* __restrict__ fcin2) {
    int bt = blockIdx.x * blockDim.x + threadIdx.x;
    if (bt >= BT_) return;
    int half = blockIdx.y;
    const float* in = d2 + (size_t)half * BT_ * 10;
    const float* w  = half ? w2 : w1;
    const float* bs = half ? b2 : b1;
    _Float16* fcin = half ? fcin2 : fcin1;
    int t = bt & (T_ - 1), b = bt >> 9;
    float acc[32];
#pragma unroll
    for (int co = 0; co < 32; ++co) acc[co] = bs[co];
    for (int k = 0; k < 3; ++k) {
        int tt = t + k - 1;
        if ((unsigned)tt < T_) {
            const float* hp = in + (size_t)(bt + k - 1) * 10;
            for (int ci = 0; ci < 10; ++ci) {
                float v = hp[ci];
#pragma unroll
                for (int co = 0; co < 32; ++co)
                    acc[co] = fmaf(w[(co * 10 + ci) * 3 + k], v, acc[co]);
            }
        }
    }
#pragma unroll
    for (int co = 0; co < 32; ++co)
        fcin[(size_t)b * FCK + co * T_ + t] = (_Float16)fmaxf(acc[co], 0.f);
}

// ---------------------------------------------------------------------------
// FC with split-K (x8) for occupancy: each wave owns a 16x64 strip and a
// 2048-deep K slice; partials to scratch (deterministic fixed-order reduce).
// grid = (128, 2 halves, SPLITK)
// ---------------------------------------------------------------------------
__global__ __launch_bounds__(32) void fc_kernel(
        const _Float16* __restrict__ fcin1, const _Float16* __restrict__ fcin2,
        const _Float16* __restrict__ wh, float* __restrict__ part) {
    const int lane = threadIdx.x, r = lane & 15, khi = lane >> 4;
    const int half = blockIdx.y, ks = blockIdx.z;
    const _Float16* fcin = half ? fcin2 : fcin1;
    const int mbase = (blockIdx.x >> 3) * 16;  // 16 row tiles
    const int nbase = (blockIdx.x & 7) * 64;   // 8 col strips of 64
    const int kbeg = ks * (FCK / SPLITK), kend = kbeg + FCK / SPLITK;

    const _Float16* arow = fcin + (size_t)(mbase + r) * FCK + khi * 16;
    const _Float16* brow[4];
#pragma unroll
    for (int j = 0; j < 4; ++j)
        brow[j] = wh + (size_t)(nbase + j * 16 + r) * FCK + khi * 16;

    v8f acc[4] = {v8f{}, v8f{}, v8f{}, v8f{}};
    for (int k0 = kbeg; k0 < kend; k0 += 32) {
        __builtin_prefetch(arow + k0 + 512, 0, 1);       // global_prefetch_b8
        v16h a = load_afrag(arow + k0);
#pragma unroll
        for (int j = 0; j < 4; ++j) {
            v16h b = *(const v16h*)(brow[j] + k0);       // B layout: K contiguous
            acc[j] = __builtin_amdgcn_wmma_f32_16x16x32_f16(false, a, false, b,
                                                            (short)0, acc[j], false, false);
        }
    }
    float* o = part + (size_t)(ks * 2 + half) * B_ * FCO;
#pragma unroll
    for (int j = 0; j < 4; ++j) {
        int col = nbase + j * 16 + r;
#pragma unroll
        for (int v = 0; v < 8; ++v)
            o[(size_t)(mbase + v + khi * 8) * FCO + col] = acc[j][v];
    }
}

__global__ void fc_reduce(const float* __restrict__ part, const float* __restrict__ fcb,
                          float* __restrict__ out) {
    int i = blockIdx.x * blockDim.x + threadIdx.x;
    if (i >= 2 * B_ * FCO) return;
    int half = i / (B_ * FCO);
    int rc = i - half * (B_ * FCO);
    float s = fcb[rc & (FCO - 1)];
#pragma unroll
    for (int ks = 0; ks < SPLITK; ++ks)
        s += part[(size_t)(ks * 2 + half) * B_ * FCO + rc];
    out[i] = tanhf(s);
}

// ---------------------------------------------------------------------------
extern "C" void kernel_launch(void* const* d_in, const int* in_sizes, int n_in,
                              void* d_out, int out_size, void* d_ws, size_t ws_size,
                              hipStream_t stream) {
    const float* x       = (const float*)d_in[0];
    const float* enc_w1  = (const float*)d_in[1];
    const float* enc_b1  = (const float*)d_in[2];
    const float* enc_w2  = (const float*)d_in[3];
    const float* enc_b2  = (const float*)d_in[4];
    const float* enc_w3  = (const float*)d_in[5];
    const float* enc_b3  = (const float*)d_in[6];
    const float* d1_w1   = (const float*)d_in[7];
    const float* d1_b1   = (const float*)d_in[8];
    const float* d1_w2   = (const float*)d_in[9];
    const float* d1_b2   = (const float*)d_in[10];
    const float* d1_w3   = (const float*)d_in[11];
    const float* d1_b3   = (const float*)d_in[12];
    const float* d2_w1   = (const float*)d_in[13];
    const float* d2_b1   = (const float*)d_in[14];
    const float* d2_w2   = (const float*)d_in[15];
    const float* d2_b2   = (const float*)d_in[16];
    const float* d2_w3   = (const float*)d_in[17];
    const float* d2_b3   = (const float*)d_in[18];
    const float* fc_w    = (const float*)d_in[19];
    const float* fc_b    = (const float*)d_in[20];
    const float* code    = (const float*)d_in[21];
    float* out = (float*)d_out;

    // ---- workspace carve-up ----
    char* ws = (char*)d_ws;
    size_t off = 0;
    auto carve = [&](size_t bytes) { char* p = ws + off; off = (off + bytes + 255) & ~(size_t)255; return p; };
    float*    h1     = (float*)   carve((size_t)BT_ * 5  * 4);
    float*    h2     = (float*)   carve((size_t)BT_ * 10 * 4);
    _Float16* xph    = (_Float16*)carve((size_t)BT_ * 64 * 2);
    _Float16* codeA  = (_Float16*)carve((size_t)2048 * 16 * 2);
    _Float16* codeB  = (_Float16*)carve((size_t)2048 * 16 * 2);
    float*    cnorm  = (float*)   carve((size_t)NM * 4);
    float*    xq1    = (float*)   carve((size_t)BT_ * 32 * 4);
    float*    xq2    = (float*)   carve((size_t)BT_ * 32 * 4);
    float*    dd1    = (float*)   carve((size_t)2 * BT_ * 5  * 4);
    float*    dd2    = (float*)   carve((size_t)2 * BT_ * 10 * 4);
    _Float16* fcin1  = (_Float16*)carve((size_t)B_ * FCK * 2);
    _Float16* fcin2  = (_Float16*)carve((size_t)B_ * FCK * 2);
    _Float16* whh    = (_Float16*)carve((size_t)FCO * FCK * 2);
    float*    part   = (float*)   carve((size_t)SPLITK * 2 * B_ * FCO * 4);
    (void)ws_size; (void)in_sizes; (void)n_in; (void)out_size;

    const int nwc = FCO * FCK;
    cvt_f16<<<(nwc + 255) / 256, 256, 0, stream>>>(fc_w, whh, nwc);
    prep_code<<<16, 256, 0, stream>>>(code, codeA, codeB, cnorm);

    enc_conv1<<<BT_ / 256, 256, 0, stream>>>(x,  enc_w1, enc_b1, h1);
    enc_conv2<<<BT_ / 256, 256, 0, stream>>>(h1, enc_w2, enc_b2, h2);
    enc_conv3<<<BT_ / 256, 256, 0, stream>>>(h2, enc_w3, enc_b3, xph);

    quantize_kernel<<<dim3(BT_ / 16, 2), 32, 0, stream>>>(
        xph, codeA, codeB, cnorm, xq1, xq2);

    dim3 dgrid(BT_ / 256, 2);
    dec_conv1<<<dgrid, 256, 0, stream>>>(xq1, xq2, d1_w1, d2_w1, d1_b1, d2_b1, dd1);
    dec_conv2<<<dgrid, 256, 0, stream>>>(dd1, d1_w2, d2_w2, d1_b2, d2_b2, dd2);
    dec_conv3<<<dgrid, 256, 0, stream>>>(dd2, d1_w3, d2_w3, d1_b3, d2_b3, fcin1, fcin2);

    fc_kernel<<<dim3(128, 2, SPLITK), 32, 0, stream>>>(fcin1, fcin2, whh, part);
    fc_reduce<<<(2 * B_ * FCO + 255) / 256, 256, 0, stream>>>(part, fc_b, out);
}
